// HeteroGNN_9981503996384
// MI455X (gfx1250) — compile-verified, matrix operand order checked by reading, not run on hardware
//
#include <hip/hip_runtime.h>
#include <hip/hip_bf16.h>
#include <math.h>

// ---------------------------------------------------------------------------
// Types for CDNA5 WMMA
// ---------------------------------------------------------------------------
typedef __attribute__((ext_vector_type(16))) __bf16          v16bf;
typedef __attribute__((ext_vector_type(16))) unsigned short  v16us;
typedef __attribute__((ext_vector_type(8)))  float           v8f;

__device__ __forceinline__ unsigned short f2bf(float f) {
    unsigned int u = __float_as_uint(f);
    u += 0x7FFFu + ((u >> 16) & 1u);          // round-to-nearest-even
    return (unsigned short)(u >> 16);
}

// ---------------------------------------------------------------------------
// GEMM: C[M x 64] = A[M x K] * W[K x 64], via v_wmma_f32_16x16x32_bf16.
// 256 threads = 8 waves; each wave computes a 16-row x 64-col tile.
// W is staged fp32->bf16 into LDS, stored transposed [n][k] so B-fragment
// K-pairs are contiguous 32-bit LDS loads (lower to ds_load_b128 pairs).
// A rows are CLAMPED (not predicated): out-of-range rows produce garbage C
// rows that are masked at store, so loads stay branch-free float4s.
// ---------------------------------------------------------------------------
__global__ __launch_bounds__(256)
void gemm_wmma_bf16(const float* __restrict__ A, const float* __restrict__ W,
                    float* __restrict__ C, int M, int K) {
    __shared__ unsigned short ldsWT[64 * 256];   // [n][k], max K=256 -> 32KB

    const int tid = threadIdx.x;
    for (int idx = tid; idx < K * 64; idx += 256) {
        int k = idx >> 6, n = idx & 63;          // W row-major [k][n]
        ldsWT[n * K + k] = f2bf(W[idx]);
    }
    __syncthreads();

    const int wave = tid >> 5, lane = tid & 31;
    const int rowbase = blockIdx.x * 128 + wave * 16;
    if (rowbase >= M) return;

    const int hi = lane >> 4;        // half-wave select
    const int ln = lane & 15;
    const int hb = hi ? 8 : 0;       // K base offset for this half-wave
    int arow = rowbase + ln;         // A row held by this lane
    if (arow >= M) arow = M - 1;     // clamp: garbage rows masked at store
    const float* Arow = A + (size_t)arow * K;

    v8f acc[4] = {v8f{}, v8f{}, v8f{}, v8f{}};

    for (int k0 = 0; k0 < K; k0 += 32) {
        // ---- A fragment: lane's K slice = [k0+hb .. +7] and [k0+hb+16 .. +23]
        const float4 p0 = *(const float4*)(Arow + k0 + hb);
        const float4 p1 = *(const float4*)(Arow + k0 + hb + 4);
        const float4 p2 = *(const float4*)(Arow + k0 + hb + 16);
        const float4 p3 = *(const float4*)(Arow + k0 + hb + 20);

        v16us au;
        au[0]  = f2bf(p0.x);  au[1]  = f2bf(p0.y);
        au[2]  = f2bf(p0.z);  au[3]  = f2bf(p0.w);
        au[4]  = f2bf(p1.x);  au[5]  = f2bf(p1.y);
        au[6]  = f2bf(p1.z);  au[7]  = f2bf(p1.w);
        au[8]  = f2bf(p2.x);  au[9]  = f2bf(p2.y);
        au[10] = f2bf(p2.z);  au[11] = f2bf(p2.w);
        au[12] = f2bf(p3.x);  au[13] = f2bf(p3.y);
        au[14] = f2bf(p3.z);  au[15] = f2bf(p3.w);
        v16bf afrag = __builtin_bit_cast(v16bf, au);

        // ---- 4 B fragments (32x16 each) from LDS, mirrored layout ----
#pragma unroll
        for (int nt = 0; nt < 4; ++nt) {
            const int n = nt * 16 + ln;
            v16us bu;
#pragma unroll
            for (int r = 0; r < 8; ++r) {
                int kl = ((r < 4) ? (2 * r) : (16 + 2 * (r - 4))) + hb;
                unsigned int pair =
                    *(const unsigned int*)&ldsWT[n * K + k0 + kl]; // (k,k+1)
                bu[2 * r]     = (unsigned short)(pair & 0xFFFFu);
                bu[2 * r + 1] = (unsigned short)(pair >> 16);
            }
            v16bf bfrag = __builtin_bit_cast(v16bf, bu);
            acc[nt] = __builtin_amdgcn_wmma_f32_16x16x32_bf16(
                false, afrag, false, bfrag, (short)0, acc[nt], false, false);
        }
    }

    // ---- store: C/D layout = VGPR r -> M = rowbase + r + 8*hi, N = lane&15
    if (rowbase + 16 <= M) {
        // interior tile: branch-free stores
#pragma unroll
        for (int nt = 0; nt < 4; ++nt) {
            const int col = nt * 16 + ln;
#pragma unroll
            for (int r = 0; r < 8; ++r) {
                int row = rowbase + r + hi * 8;
                C[(size_t)row * 64 + col] = acc[nt][r];
            }
        }
    } else {
#pragma unroll
        for (int nt = 0; nt < 4; ++nt) {
            const int col = nt * 16 + ln;
#pragma unroll
            for (int r = 0; r < 8; ++r) {
                int row = rowbase + r + hi * 8;
                if (row < M) C[(size_t)row * 64 + col] = acc[nt][r];
            }
        }
    }
}

// ---------------------------------------------------------------------------
// Small helper kernels
// ---------------------------------------------------------------------------
__global__ void fill_kernel(float* __restrict__ p, float v, int n) {
    int i = blockIdx.x * blockDim.x + threadIdx.x;
    if (i < n) p[i] = v;
}

__global__ void dot64_kernel(const float* __restrict__ Hm,
                             const float* __restrict__ a,
                             float* __restrict__ out, int Nn) {
    int n = blockIdx.x * blockDim.x + threadIdx.x;
    if (n >= Nn) return;
    const float* row = Hm + (size_t)n * 64;
    float s = 0.0f;
#pragma unroll
    for (int f = 0; f < 64; ++f) s += row[f] * a[f];
    out[n] = s;
}

__global__ void bias_init_kernel(float* __restrict__ x, int Nn,
                                 const float* b0, const float* b1,
                                 const float* b2) {
    int idx = blockIdx.x * blockDim.x + threadIdx.x;
    if (idx >= Nn * 64) return;
    int f = idx & 63;
    float v = 0.0f;
    if (b0) v += b0[f];
    if (b1) v += b1[f];
    if (b2) v += b2[f];
    x[idx] = v;
}

__device__ __forceinline__ void atomicMaxF(float* addr, float v) {
    int* ai = (int*)addr;
    int old = __float_as_int(*addr);
    while (__int_as_float(old) < v) {
        int assumed = old;
        old = atomicCAS(ai, assumed, __float_as_int(v));
        if (old == assumed) break;
    }
}

__global__ void edge_logit_max_kernel(const int* __restrict__ src,
                                      const int* __restrict__ dst,
                                      const float* __restrict__ ls,
                                      const float* __restrict__ ld,
                                      float* __restrict__ logits,
                                      float* __restrict__ m, int E) {
    int e = blockIdx.x * blockDim.x + threadIdx.x;
    if (e >= E) return;
    float lg = ls[src[e]] + ld[dst[e]];
    lg = lg > 0.0f ? lg : 0.2f * lg;             // leaky_relu(0.2)
    logits[e] = lg;
    atomicMaxF(&m[dst[e]], lg);
}

__global__ void edge_exp_den_kernel(const int* __restrict__ dst,
                                    float* __restrict__ logits,
                                    const float* __restrict__ m,
                                    float* __restrict__ den, int E) {
    int e = blockIdx.x * blockDim.x + threadIdx.x;
    if (e >= E) return;
    int d = dst[e];
    float ex = __expf(logits[e] - m[d]);
    logits[e] = ex;                               // reuse buffer for ex
    atomicAdd(&den[d], ex);
}

__global__ void edge_scatter_kernel(const int* __restrict__ src,
                                    const int* __restrict__ dst,
                                    const float* __restrict__ ex,
                                    const float* __restrict__ den,
                                    const float* __restrict__ hs,
                                    float* __restrict__ out, int E) {
    int t = blockIdx.x * blockDim.x + threadIdx.x;
    int e = t >> 6;
    if (e >= E) return;
    int f = t & 63;
    int d = dst[e];
    float alpha = ex[e] / (den[d] + 1e-16f);
    atomicAdd(&out[(size_t)d * 64 + f], hs[(size_t)src[e] * 64 + f] * alpha);
}

__global__ void relu_kernel(float* __restrict__ x, int n) {
    int i = blockIdx.x * blockDim.x + threadIdx.x;
    if (i < n) x[i] = x[i] > 0.0f ? x[i] : 0.0f;
}

__global__ void pool_scatter_kernel(const float* __restrict__ x,
                                    const int* __restrict__ batch,
                                    float* __restrict__ pooled,
                                    float* __restrict__ cnt, int Nn) {
    int idx = blockIdx.x * blockDim.x + threadIdx.x;
    if (idx >= Nn * 64) return;
    int n = idx >> 6, f = idx & 63;
    int g = batch[n];
    atomicAdd(&pooled[(size_t)g * 64 + f], x[idx]);
    if (f == 0) atomicAdd(&cnt[g], 1.0f);
}

__global__ void final_kernel(const float* __restrict__ pooled,  // [5][G][64]
                             const float* __restrict__ cnt,     // [5][G]
                             const float* __restrict__ post_emb,// [G][768]
                             const float* __restrict__ Wlin,    // [1088][2]
                             const float* __restrict__ blin,    // [2]
                             float* __restrict__ out) {         // [G][2]
    int g = blockIdx.x * blockDim.x + threadIdx.x;
    if (g >= 1024) return;
    float a0 = blin[0], a1 = blin[1];
    int k = 0;
    for (int t = 0; t < 5; ++t) {
        float c = cnt[t * 1024 + g];
        c = c > 1.0f ? c : 1.0f;
        float inv = 1.0f / c;
        const float* pr = pooled + ((size_t)t * 1024 + g) * 64;
        for (int f = 0; f < 64; ++f, ++k) {
            float v = pr[f] * inv;
            a0 += v * Wlin[k * 2 + 0];
            a1 += v * Wlin[k * 2 + 1];
        }
    }
    const float* pe = post_emb + (size_t)g * 768;
    for (int p = 0; p < 768; ++p, ++k) {
        float v = pe[p];
        a0 += v * Wlin[k * 2 + 0];
        a1 += v * Wlin[k * 2 + 1];
    }
    a0 = a0 > 0.0f ? a0 : 0.0f;
    a1 = a1 > 0.0f ? a1 : 0.0f;
    float mx = a0 > a1 ? a0 : a1;
    float e0 = __expf(a0 - mx), e1 = __expf(a1 - mx);
    float s = e0 + e1;
    out[g * 2 + 0] = e0 / s;
    out[g * 2 + 1] = e1 / s;
}

// ---------------------------------------------------------------------------
// Host orchestration
// ---------------------------------------------------------------------------
extern "C" void kernel_launch(void* const* d_in, const int* in_sizes, int n_in,
                              void* d_out, int out_size, void* d_ws,
                              size_t ws_size, hipStream_t stream) {
    (void)in_sizes; (void)n_in; (void)out_size; (void)ws_size;

    // node types: 0=tag 1=question 2=answer 3=comment 4=module
    static const int NT_N[5]   = {5000, 50000, 100000, 150000, 2000};
    static const int NT_OFF[5] = {0, 5000, 55000, 155000, 305000};
    static const int ET_S[10]  = {0, 0, 0, 4, 4, 1, 2, 3, 1, 2};
    static const int ET_D[10]  = {1, 2, 3, 1, 2, 0, 0, 0, 4, 4};
    static const int ET_E[10]  = {500000, 1000000, 1000000, 300000, 600000,
                                  500000, 1000000, 1000000, 300000, 600000};
    // incoming edge-type ids per dst node type (-1 = none)
    static const int INC[5][3] = {{5, 6, 7}, {0, 3, -1}, {1, 4, -1},
                                  {2, -1, -1}, {8, 9, -1}};
    const int NTOT = 307000;

    // workspace layout (floats)
    float* ws    = (float*)d_ws;
    float* bufA  = ws;                            // NTOT*64 (layer-0 output)
    float* bufB  = bufA + (size_t)NTOT * 64;      // NTOT*64 (layer-1 output)
    float* hs    = bufB + (size_t)NTOT * 64;      // 150000*64
    float* hd    = hs   + (size_t)150000 * 64;    // 150000*64
    float* lsb   = hd   + (size_t)150000 * 64;    // 150000
    float* ldb   = lsb  + 150000;                 // 150000
    float* mbuf  = ldb  + 150000;                 // 150000
    float* den   = mbuf + 150000;                 // 150000
    float* ebuf  = den  + 150000;                 // 1,000,000
    float* pooled= ebuf + 1000000;                // 5*1024*64
    float* cnt   = pooled + 5 * 1024 * 64;        // 5*1024

    const float NEG_INF = -3.402823466e+38f;

    for (int layer = 0; layer < 2; ++layer) {
        const float* Wsrc = (const float*)d_in[layer ? 26 : 21];
        const float* Wdst = (const float*)d_in[layer ? 27 : 22];
        const float* asrc = (const float*)d_in[layer ? 28 : 23];
        const float* adst = (const float*)d_in[layer ? 29 : 24];
        const float* bb   = (const float*)d_in[layer ? 30 : 25];
        const int K = layer ? 64 : 256;
        float* xout = layer ? bufB : bufA;

        // init output accumulators with per-dst-type bias sums
        for (int t = 0; t < 5; ++t) {
            const float* b0 = (INC[t][0] >= 0) ? bb + INC[t][0] * 64 : nullptr;
            const float* b1 = (INC[t][1] >= 0) ? bb + INC[t][1] * 64 : nullptr;
            const float* b2 = (INC[t][2] >= 0) ? bb + INC[t][2] * 64 : nullptr;
            int n = NT_N[t] * 64;
            bias_init_kernel<<<(n + 255) / 256, 256, 0, stream>>>(
                xout + (size_t)NT_OFF[t] * 64, NT_N[t], b0, b1, b2);
        }

        for (int i = 0; i < 10; ++i) {
            const int s = ET_S[i], d = ET_D[i];
            const int Ns = NT_N[s], Nd = NT_N[d], E = ET_E[i];
            const float* As = layer ? bufA + (size_t)NT_OFF[s] * 64
                                    : (const float*)d_in[s];
            const float* Ad = layer ? bufA + (size_t)NT_OFF[d] * 64
                                    : (const float*)d_in[d];
            const int* ei  = (const int*)d_in[6 + i];
            const int* src = ei;
            const int* dst = ei + E;

            gemm_wmma_bf16<<<(Ns + 127) / 128, 256, 0, stream>>>(
                As, Wsrc + (size_t)i * K * 64, hs, Ns, K);
            gemm_wmma_bf16<<<(Nd + 127) / 128, 256, 0, stream>>>(
                Ad, Wdst + (size_t)i * K * 64, hd, Nd, K);

            dot64_kernel<<<(Ns + 255) / 256, 256, 0, stream>>>(
                hs, asrc + i * 64, lsb, Ns);
            dot64_kernel<<<(Nd + 255) / 256, 256, 0, stream>>>(
                hd, adst + i * 64, ldb, Nd);

            fill_kernel<<<(Nd + 255) / 256, 256, 0, stream>>>(mbuf, NEG_INF, Nd);
            fill_kernel<<<(Nd + 255) / 256, 256, 0, stream>>>(den, 0.0f, Nd);

            edge_logit_max_kernel<<<(E + 255) / 256, 256, 0, stream>>>(
                src, dst, lsb, ldb, ebuf, mbuf, E);
            edge_exp_den_kernel<<<(E + 255) / 256, 256, 0, stream>>>(
                dst, ebuf, mbuf, den, E);
            edge_scatter_kernel<<<(E * 64 + 255) / 256, 256, 0, stream>>>(
                src, dst, ebuf, den, hs, xout + (size_t)NT_OFF[d] * 64, E);
        }

        relu_kernel<<<(NTOT * 64 + 255) / 256, 256, 0, stream>>>(
            xout, NTOT * 64);
    }

    // -------- pooling + head --------
    fill_kernel<<<(5 * 1024 * 64 + 255) / 256, 256, 0, stream>>>(
        pooled, 0.0f, 5 * 1024 * 64);
    fill_kernel<<<(5 * 1024 + 255) / 256, 256, 0, stream>>>(
        cnt, 0.0f, 5 * 1024);
    for (int t = 0; t < 5; ++t) {
        int n = NT_N[t] * 64;
        pool_scatter_kernel<<<(n + 255) / 256, 256, 0, stream>>>(
            bufB + (size_t)NT_OFF[t] * 64, (const int*)d_in[16 + t],
            pooled + (size_t)t * 1024 * 64, cnt + t * 1024, NT_N[t]);
    }
    final_kernel<<<4, 256, 0, stream>>>(
        pooled, cnt, (const float*)d_in[5], (const float*)d_in[31],
        (const float*)d_in[32], (float*)d_out);
}
